// Attention_27547920236796
// MI455X (gfx1250) — compile-verified, matrix operand order checked by reading, not run on hardware
//
#include <hip/hip_runtime.h>

// ---------------------------------------------------------------------------
// Causal single-head attention for MI455X (gfx1250), wave32 + WMMA bf16.
//   x:[4,2048,1024]f32, wq/wk/wv:[1024,128]f32, wo:[128,1024]f32
//   out:[4,2048,1024]f32
// ---------------------------------------------------------------------------

typedef __bf16 bf16;
typedef __attribute__((ext_vector_type(16))) __bf16 v16bf;
typedef __attribute__((ext_vector_type(8)))  __bf16 v8bf;
typedef __attribute__((ext_vector_type(8)))  float  v8f;

#define B_SZ   4
#define S_SZ   2048
#define DM     1024
#define DH     128
#define NEG_INF (-__builtin_inff())

#define WMMA_BF16(a, b, c) \
  __builtin_amdgcn_wmma_f32_16x16x32_bf16(false, (a), false, (b), (short)0, (c), false, false)

static __device__ __forceinline__ v16bf cat8(v8bf lo, v8bf hi) {
  return __builtin_shufflevector(lo, hi, 0,1,2,3,4,5,6,7,8,9,10,11,12,13,14,15);
}
static __device__ __forceinline__ v8bf ld8(const bf16* p) { return *(const v8bf*)p; }
static __device__ __forceinline__ v8bf cvt8(float4 a, float4 b) {
  v8bf r;
  r[0]=(bf16)a.x; r[1]=(bf16)a.y; r[2]=(bf16)a.z; r[3]=(bf16)a.w;
  r[4]=(bf16)b.x; r[5]=(bf16)b.y; r[6]=(bf16)b.z; r[7]=(bf16)b.w;
  return r;
}
static __device__ __forceinline__ v8f zero8() {
  v8f z;
  #pragma unroll
  for (int i = 0; i < 8; ++i) z[i] = 0.f;
  return z;
}

// --------------------- gfx1250 async global->LDS copy ----------------------
#if defined(__HIP_DEVICE_COMPILE__) && \
    __has_builtin(__builtin_amdgcn_global_load_async_to_lds_b128)
#define HAVE_ASYNC_LDS 1
#endif

// The builtin's params are GCC-style int4 vectors in AS1 (global) / AS3 (LDS):
//   void __builtin_amdgcn_global_load_async_to_lds_b128(v4i AS1*, v4i AS3*, imm, imm)
typedef int v4i __attribute__((vector_size(16)));
typedef __attribute__((address_space(1))) v4i gv4i;
typedef __attribute__((address_space(3))) v4i lv4i;

// Copy 16 bytes global -> LDS. Async (ASYNCcnt) when available, else via VGPR.
static __device__ __forceinline__ void cp16_g2l(void* lds, const void* g) {
#ifdef HAVE_ASYNC_LDS
  __builtin_amdgcn_global_load_async_to_lds_b128(
      (gv4i*)g, (lv4i*)lds, 0, 0);
#else
  *(uint4*)lds = *(const uint4*)g;
#endif
}
static __device__ __forceinline__ void cp_wait() {
#ifdef HAVE_ASYNC_LDS
  asm volatile("s_wait_asynccnt 0x0" ::: "memory");
#endif
}

// ---------------------------------------------------------------------------
// Kernel 0: weight transpose + fp32 -> bf16.  wqT/wkT/wvT:[128][1024] (B-operand
// layout: lane=column N, contiguous K).  woT:[1024][128].  1/sqrt(128) folded
// into wqT.
// ---------------------------------------------------------------------------
__global__ void __launch_bounds__(256) prep_w(
    const float* __restrict__ wq, const float* __restrict__ wk,
    const float* __restrict__ wv, const float* __restrict__ wo,
    bf16* __restrict__ wqT, bf16* __restrict__ wkT,
    bf16* __restrict__ wvT, bf16* __restrict__ woT) {
  int i = blockIdx.x * blockDim.x + threadIdx.x;     // 0 .. 1024*128-1
  if (i >= DM * DH) return;
  int k = i >> 7;        // 0..1023
  int n = i & 127;       // 0..127
  wqT[n * DM + k] = (bf16)(wq[i] * 0.08838834764831845f);  // * 1/sqrt(128)
  wkT[n * DM + k] = (bf16)wk[i];
  wvT[n * DM + k] = (bf16)wv[i];
  woT[k * DH + n] = (bf16)wo[n * DM + k];                  // woT:[1024 N][128 K]
}

// ---------------------------------------------------------------------------
// Kernel 1: QKV projection.  blockIdx.y selects which of {Q,K,V} this block
// computes (3x parallelism; x rows stay L2-hot across the three).  One wave
// per 16-row M-tile, 8 accumulators (16x128 strip), K-loop of 32.
// V written transposed: vT[b][128][2048].
// ---------------------------------------------------------------------------
__global__ void __launch_bounds__(128) qkv_proj(
    const float* __restrict__ x,
    const bf16* __restrict__ wqT, const bf16* __restrict__ wkT,
    const bf16* __restrict__ wvT,
    bf16* __restrict__ qb, bf16* __restrict__ kbuf, bf16* __restrict__ vT) {
  const int lane = threadIdx.x & 31;
  const int wave = threadIdx.x >> 5;
  const int l16  = lane & 15;
  const int half = lane >> 4;
  const int w    = blockIdx.y;              // 0=Q, 1=K, 2=V
  const int mt   = blockIdx.x * 4 + wave;   // 0..511
  const int row0 = mt * 16;
  const int bidx = row0 >> 11;
  const int s0   = row0 & (S_SZ - 1);

  const float* xrow = x + (size_t)(row0 + l16) * DM;
  const bf16* wT = (w == 0) ? wqT : ((w == 1) ? wkT : wvT);

  v8f acc[8];
  #pragma unroll
  for (int nt = 0; nt < 8; ++nt) acc[nt] = zero8();

  for (int k0 = 0; k0 < DM; k0 += 32) {
    const float4* pa = (const float4*)(xrow + k0 + half * 8);
    const float4* pc = (const float4*)(xrow + k0 + 16 + half * 8);
    v16bf afr = cat8(cvt8(pa[0], pa[1]), cvt8(pc[0], pc[1]));
    #pragma unroll
    for (int nt = 0; nt < 8; ++nt) {
      const bf16* wp = wT + (size_t)(nt * 16 + l16) * DM + k0 + half * 8;
      v16bf bfr = cat8(ld8(wp), ld8(wp + 16));
      acc[nt] = WMMA_BF16(afr, bfr, acc[nt]);
    }
  }

  if (w < 2) {
    bf16* dst = (w == 0) ? qb : kbuf;
    #pragma unroll
    for (int nt = 0; nt < 8; ++nt)
      #pragma unroll
      for (int r = 0; r < 8; ++r)
        dst[(size_t)(row0 + r + half * 8) * DH + nt * 16 + l16] = (bf16)acc[nt][r];
  } else {
    bf16* dst = vT + (size_t)bidx * DH * S_SZ;
    #pragma unroll
    for (int nt = 0; nt < 8; ++nt)
      #pragma unroll
      for (int r = 0; r < 8; ++r)
        dst[(size_t)(nt * 16 + l16) * S_SZ + (s0 + r + half * 8)] = (bf16)acc[nt][r];
  }
}

// ---------------------------------------------------------------------------
// Kernel 2: flash attention (causal).  Block = 4 waves = 64 queries; 64-key
// K/V^T tiles double-buffered in LDS via async global->LDS copies (ASYNCcnt),
// prefetching tile ib+1 while computing tile ib.  Q fragments live in
// registers for the whole kernel.
// ---------------------------------------------------------------------------
__global__ void __launch_bounds__(128) flash_attn(
    const bf16* __restrict__ qb, const bf16* __restrict__ kbuf,
    const bf16* __restrict__ vT, bf16* __restrict__ ctx) {
  __shared__ __align__(16) bf16 Ks[2][64][DH];   // 2 x 16 KB [key][dim]
  __shared__ __align__(16) bf16 Vs[2][DH][64];   // 2 x 16 KB [dim][key] (V^T)
  __shared__ __align__(16) bf16 Ps[4][16][64];   //     8 KB per-wave P scratch

  const int lane = threadIdx.x & 31;
  const int wave = threadIdx.x >> 5;
  const int l16  = lane & 15;
  const int half = lane >> 4;
  const int bidx = blockIdx.y;
  const int qb0  = blockIdx.x * 64;
  const int q0   = qb0 + wave * 16;
  const int tid  = threadIdx.x;

  const bf16* kbase = kbuf + (size_t)bidx * S_SZ * DH;
  const bf16* vbase = vT + (size_t)bidx * DH * S_SZ;

  // Preload Q fragments (16 queries x 128 dims -> 4 chunks of K=32)
  v16bf qf[4];
  {
    const bf16* qrow = qb + (size_t)(bidx * S_SZ + q0 + l16) * DH;
    #pragma unroll
    for (int c = 0; c < 4; ++c) {
      const bf16* p = qrow + c * 32 + half * 8;
      qf[c] = cat8(ld8(p), ld8(p + 16));
    }
  }

  float m_i[8], l_i[8];
  v8f o[8];
  #pragma unroll
  for (int r = 0; r < 8; ++r) { m_i[r] = NEG_INF; l_i[r] = 0.f; }
  #pragma unroll
  for (int nt = 0; nt < 8; ++nt) o[nt] = zero8();

  const int nkb = (qb0 >> 6) + 1;   // key blocks of 64, causal limit for block

  // Stage tile 0 into buffer 0.
  {
    const uint4* src = (const uint4*)kbase;          // k0 = 0
    for (int i = tid; i < 1024; i += 128)
      cp16_g2l(&((uint4*)&Ks[0][0][0])[i], &src[i]);
    for (int i = tid; i < 1024; i += 128) {
      int row = i >> 3, ch = i & 7;
      cp16_g2l(&Vs[0][row][ch * 8], vbase + (size_t)row * S_SZ + ch * 8);
    }
  }

  for (int ib = 0; ib < nkb; ++ib) {
    const int k0  = ib * 64;
    const int buf = ib & 1;
    cp_wait();            // this wave's async copies for tile ib are done
    __syncthreads();      // everyone's copies done; previous compute done

    // Prefetch tile ib+1 into the other buffer while we compute on `buf`.
    if (ib + 1 < nkb) {
      const int kn = k0 + 64;
      const uint4* src = (const uint4*)(kbase + (size_t)kn * DH);
      for (int i = tid; i < 1024; i += 128)
        cp16_g2l(&((uint4*)&Ks[buf ^ 1][0][0])[i], &src[i]);
      for (int i = tid; i < 1024; i += 128) {
        int row = i >> 3, ch = i & 7;
        cp16_g2l(&Vs[buf ^ 1][row][ch * 8],
                 vbase + (size_t)row * S_SZ + kn + ch * 8);
      }
    }

    if (k0 <= q0 + 15) {   // wave not fully masked (uniform per wave)
      // Scores: S = Q K^T  (4 subtiles of 16 keys)
      v8f sc[4];
      #pragma unroll
      for (int st = 0; st < 4; ++st) sc[st] = zero8();
      #pragma unroll
      for (int c = 0; c < 4; ++c) {
        #pragma unroll
        for (int st = 0; st < 4; ++st) {
          const bf16* p = &Ks[buf][st * 16 + l16][c * 32 + half * 8];
          v16bf kf = cat8(ld8(p), ld8(p + 16));
          sc[st] = WMMA_BF16(qf[c], kf, sc[st]);
        }
      }
      // Causal mask (1/sqrt(dh) already folded into Q)
      #pragma unroll
      for (int st = 0; st < 4; ++st) {
        const int key = k0 + st * 16 + l16;
        #pragma unroll
        for (int r = 0; r < 8; ++r)
          if (key > q0 + r + half * 8) sc[st][r] = NEG_INF;
      }
      // Online softmax: row lives across the 16 lanes of this half, element r.
      #pragma unroll
      for (int r = 0; r < 8; ++r) {
        float mx = fmaxf(fmaxf(sc[0][r], sc[1][r]), fmaxf(sc[2][r], sc[3][r]));
        #pragma unroll
        for (int d = 1; d < 16; d <<= 1) mx = fmaxf(mx, __shfl_xor(mx, d, 32));
        const float mnew  = fmaxf(m_i[r], mx);
        const float alpha = __expf(m_i[r] - mnew);
        float rs = 0.f;
        #pragma unroll
        for (int st = 0; st < 4; ++st) {
          float p = __expf(sc[st][r] - mnew);
          sc[st][r] = p;
          rs += p;
        }
        #pragma unroll
        for (int d = 1; d < 16; d <<= 1) rs += __shfl_xor(rs, d, 32);
        l_i[r] = l_i[r] * alpha + rs;
        m_i[r] = mnew;
        #pragma unroll
        for (int nt = 0; nt < 8; ++nt) o[nt][r] *= alpha;
      }
      // Accumulator layout -> A-fragment layout via per-wave LDS scratch.
      #pragma unroll
      for (int st = 0; st < 4; ++st)
        #pragma unroll
        for (int r = 0; r < 8; ++r)
          Ps[wave][r + half * 8][st * 16 + l16] = (bf16)sc[st][r];
      // O += P V
      #pragma unroll
      for (int kc = 0; kc < 2; ++kc) {
        const bf16* pp = &Ps[wave][l16][kc * 32 + half * 8];
        v16bf pf = cat8(ld8(pp), ld8(pp + 16));
        #pragma unroll
        for (int nt = 0; nt < 8; ++nt) {
          const bf16* vp = &Vs[buf][nt * 16 + l16][kc * 32 + half * 8];
          v16bf vf = cat8(ld8(vp), ld8(vp + 16));
          o[nt] = WMMA_BF16(pf, vf, o[nt]);
        }
      }
    }
  }

  // Normalize and store ctx (bf16, [b][s][128]).
  float inv[8];
  #pragma unroll
  for (int r = 0; r < 8; ++r) inv[r] = 1.0f / l_i[r];
  #pragma unroll
  for (int nt = 0; nt < 8; ++nt)
    #pragma unroll
    for (int r = 0; r < 8; ++r)
      ctx[(size_t)(bidx * S_SZ + q0 + r + half * 8) * DH + nt * 16 + l16] =
          (bf16)(o[nt][r] * inv[r]);
}

// ---------------------------------------------------------------------------
// Kernel 3: output projection  ctx[8192x128] * wo[128x1024] -> out f32.
// Each wave: 16 rows x 128 cols (8 accumulators), K-loop of 4 x 32.
// ---------------------------------------------------------------------------
__global__ void __launch_bounds__(256) out_proj(
    const bf16* __restrict__ ctx, const bf16* __restrict__ woT,
    float* __restrict__ out) {
  const int lane = threadIdx.x & 31;
  const int wave = threadIdx.x >> 5;
  const int l16  = lane & 15;
  const int half = lane >> 4;
  const int g    = blockIdx.x * 8 + wave;   // 0..4095
  const int mt   = g >> 3;
  const int nc0  = (g & 7) * 128;
  const int row0 = mt * 16;

  v8f acc[8];
  #pragma unroll
  for (int nt = 0; nt < 8; ++nt) acc[nt] = zero8();

  const bf16* arow = ctx + (size_t)(row0 + l16) * DH;
  #pragma unroll
  for (int c = 0; c < 4; ++c) {
    const bf16* pa = arow + c * 32 + half * 8;
    v16bf afr = cat8(ld8(pa), ld8(pa + 16));
    #pragma unroll
    for (int nt = 0; nt < 8; ++nt) {
      const bf16* pb = woT + (size_t)(nc0 + nt * 16 + l16) * DH + c * 32 + half * 8;
      v16bf bfr = cat8(ld8(pb), ld8(pb + 16));
      acc[nt] = WMMA_BF16(afr, bfr, acc[nt]);
    }
  }
  #pragma unroll
  for (int nt = 0; nt < 8; ++nt)
    #pragma unroll
    for (int r = 0; r < 8; ++r)
      out[(size_t)(row0 + r + half * 8) * DM + nc0 + nt * 16 + l16] = acc[nt][r];
}

// ---------------------------------------------------------------------------
// Launch
// ---------------------------------------------------------------------------
extern "C" void kernel_launch(void* const* d_in, const int* in_sizes, int n_in,
                              void* d_out, int out_size, void* d_ws, size_t ws_size,
                              hipStream_t stream) {
  const float* x  = (const float*)d_in[0];
  const float* wq = (const float*)d_in[1];
  const float* wk = (const float*)d_in[2];
  const float* wv = (const float*)d_in[3];
  const float* wo = (const float*)d_in[4];
  float* out = (float*)d_out;

  // Workspace layout (bf16 elements): 4 transposed weights + q/k/vT/ctx.
  bf16* W   = (bf16*)d_ws;
  bf16* wqT = W;
  bf16* wkT = wqT + (size_t)DH * DM;
  bf16* wvT = wkT + (size_t)DH * DM;
  bf16* woT = wvT + (size_t)DH * DM;
  bf16* qb  = woT + (size_t)DM * DH;
  bf16* kb  = qb  + (size_t)B_SZ * S_SZ * DH;
  bf16* vT  = kb  + (size_t)B_SZ * S_SZ * DH;
  bf16* cx  = vT  + (size_t)B_SZ * S_SZ * DH;
  // total: 4*128K + 4*1M bf16 elems ~ 9.4 MB

  prep_w<<<dim3((DM * DH + 255) / 256), 256, 0, stream>>>(
      wq, wk, wv, wo, wqT, wkT, wvT, woT);
  qkv_proj<<<dim3(128, 3), 128, 0, stream>>>(x, wqT, wkT, wvT, qb, kb, vT);
  flash_attn<<<dim3(S_SZ / 64, B_SZ), 128, 0, stream>>>(qb, kb, vT, cx);
  out_proj<<<dim3(512), 256, 0, stream>>>(cx, woT, out);
}
